// DGCNN_DA_46901042873042
// MI455X (gfx1250) — matrix operand order repaired and checked by reading, not compiled, not run on hardware
//
#include <hip/hip_runtime.h>
#include <hip/hip_bf16.h>

// ---------------------------------------------------------------------------
// DGCNN forward for MI455X (gfx1250, wave32).
//  - knn inner products: exact-f32 WMMA v_wmma_f32_16x16x4_f32, 16x64 strip/wave
//  - convs / FCs:        f16 WMMA v_wmma_f32_16x16x32_f16, f32 accumulate,
//                        multi-N-tile per wave, b128 A-fragment loads
//  - BN is training-mode (batch stats) -> separate reduction kernels
// ---------------------------------------------------------------------------

typedef _Float16 half16 __attribute__((ext_vector_type(16)));
typedef _Float16 half8  __attribute__((ext_vector_type(8)));
typedef float    f32x8  __attribute__((ext_vector_type(8)));
typedef float    f32x2  __attribute__((ext_vector_type(2)));

#define NPTS   1024
#define BATCH  32
#define KNN_K  20
#define BN_EPS 1e-5f
#define SLOPE  0.2f
#define NEGINF (-3.4e38f)

__device__ __forceinline__ float lrelu(float x) { return x > 0.f ? x : SLOPE * x; }

// ---------------------------------------------------------------------------
// knn: pd[b][n][m] = 2*<x_n,x_m> - |x_n|^2 - |x_m|^2
// One wave computes a 16 x (16*CT) strip: A fragment reused across CT tiles.
// Requires C % 4 == 0 (caller pads stage-1 xyz to C=4).
// x layout: ptr + b*bStride + (chanOff+c)*NPTS + point
// ---------------------------------------------------------------------------
template<int CT>
__global__ __launch_bounds__(32)
void knn_pd_kernel(const float* __restrict__ x, int chanOff, int C,
                   long long bStride, float* __restrict__ pd)
{
    const int lane = threadIdx.x;
    const int hi   = lane >> 4;          // half-wave select
    const int lid  = lane & 15;
    const int tN0  = blockIdx.x * CT;    // first column tile (neighbor m)
    const int tM   = blockIdx.y;         // row tile (query n)
    const int b    = blockIdx.z;
    const float* xb = x + (long long)b * bStride + (long long)chanOff * NPTS;
    const int rowBase = tM * 16;
    const int colBase = tN0 * 16;

    __shared__ float sxxA[16];        // |x|^2 of the 16 query points
    __shared__ float sxxB[CT * 16];   // |x|^2 of the CT*16 neighbor points
    for (int t = lane; t < 16 + CT * 16; t += 32) {
        int p = (t < 16) ? (rowBase + t) : (colBase + (t - 16));
        float s = 0.f;
        for (int c = 0; c < C; ++c) { float v = xb[(long long)c * NPTS + p]; s += v * v; }
        if (t < 16) sxxA[t] = s; else sxxB[t - 16] = s;
    }
    __syncthreads();

    f32x8 acc[CT];
    #pragma unroll
    for (int t = 0; t < CT; ++t) acc[t] = f32x8{0.f,0.f,0.f,0.f,0.f,0.f,0.f,0.f};

    for (int kk = 0; kk < C; kk += 4) {
        const int ka = kk + hi * 2;      // f32 A 16x4: V0=K0/K2, V1=K1/K3
        const float* r0 = xb + (long long)ka * NPTS;
        const float* r1 = xb + (long long)(ka + 1) * NPTS;
        f32x2 a;
        a[0] = r0[rowBase + lid];
        a[1] = r1[rowBase + lid];
        #pragma unroll
        for (int t = 0; t < CT; ++t) {
            f32x2 bm;
            bm[0] = r0[colBase + t * 16 + lid];
            bm[1] = r1[colBase + t * 16 + lid];
            acc[t] = __builtin_amdgcn_wmma_f32_16x16x4_f32(false, a, false, bm,
                                                           (short)0, acc[t], false, false);
        }
    }

    #pragma unroll
    for (int t = 0; t < CT; ++t) {
        float xxc = sxxB[t * 16 + lid];
        float* out = pd + ((long long)b * NPTS + rowBase) * NPTS + colBase + t * 16 + lid;
        #pragma unroll
        for (int r = 0; r < 8; ++r) {
            int m = r + hi * 8;          // D: VGPR r -> row r (lanes0-15) / r+8
            out[(long long)m * NPTS] = 2.f * acc[t][r] - sxxA[m] - xxc;
        }
    }
}

// ---------------------------------------------------------------------------
// top-20 per pd row; one wave per row; stable (smallest index wins ties)
// ---------------------------------------------------------------------------
__global__ __launch_bounds__(32)
void topk_kernel(const float* __restrict__ pd, int* __restrict__ idx)
{
    const int row  = blockIdx.x;                 // b*NPTS + n
    const int lane = threadIdx.x;
    const float* r = pd + (long long)row * NPTS;
    float v[32];
    #pragma unroll
    for (int j = 0; j < 32; ++j) v[j] = r[j * 32 + lane];
    unsigned sel = 0u;                           // per-lane selected mask
    for (int t = 0; t < KNN_K; ++t) {
        float bv = NEGINF; int bi = 0x7fffffff;
        #pragma unroll
        for (int j = 0; j < 32; ++j) {
            if ((sel >> j) & 1u) continue;
            int ei = j * 32 + lane;
            if (v[j] > bv || (v[j] == bv && ei < bi)) { bv = v[j]; bi = ei; }
        }
        for (int off = 16; off; off >>= 1) {
            float ov = __shfl_xor(bv, off, 32);
            int   oi = __shfl_xor(bi, off, 32);
            if (ov > bv || (ov == bv && oi < bi)) { bv = ov; bi = oi; }
        }
        if ((bi & 31) == lane) sel |= 1u << (bi >> 5);
        if (lane == 0) idx[(long long)row * KNN_K + t] = bi;
    }
}

// ---------------------------------------------------------------------------
// f16 WMMA GEMM: out[b](O x Ncols) = W(O x Kdim) * X[b](Kdim x Ncols) (+bias)
// One wave per 16 x (16*NT) output strip; A fragment loaded as 2x b128 and
// reused across NT tiles. Kdim % 32 == 0, Ncols % (16*NT) == 0.
// ---------------------------------------------------------------------------
template<int NT>
__global__ __launch_bounds__(32)
void gemm_f16_kernel(const _Float16* __restrict__ W, int Kdim,
                     const _Float16* __restrict__ X, int chanOff, long long xbStride,
                     const float* __restrict__ bias,
                     float* __restrict__ out, int O, int Ncols)
{
    const int lane = threadIdx.x;
    const int hi   = lane >> 4;
    const int lid  = lane & 15;
    const int tN0  = blockIdx.x * NT;
    const int tO   = blockIdx.y;
    const int b    = blockIdx.z;
    const int hiofs = hi ? 8 : 0;        // A K-pattern: runs [hiofs..+7], [16+hiofs..+7]
    const int bofs  = hi ? 8 : 0;        // B K-pattern: runs [bofs..+7], [16+bofs..+7]
    const _Float16* Wr = W + (long long)(tO * 16 + lid) * Kdim;
    const _Float16* Xb = X + (long long)b * xbStride
                           + (long long)chanOff * Ncols + tN0 * 16 + lid;

    f32x8 acc[NT];
    #pragma unroll
    for (int t = 0; t < NT; ++t) acc[t] = f32x8{0.f,0.f,0.f,0.f,0.f,0.f,0.f,0.f};

    for (int kk = 0; kk < Kdim; kk += 32) {
        half8 alo = *(const half8*)(Wr + kk + hiofs);        // global_load_b128
        half8 ahi = *(const half8*)(Wr + kk + 16 + hiofs);   // global_load_b128
        half16 a;
        #pragma unroll
        for (int h = 0; h < 8; ++h) { a[h] = alo[h]; a[h + 8] = ahi[h]; }
        if (kk + 32 < Kdim) {
            __builtin_prefetch(&Wr[kk + 32], 0, 1);                        // global_prefetch_b8
            __builtin_prefetch(&Xb[(long long)(kk + 32) * Ncols], 0, 1);
        }
        #pragma unroll
        for (int t = 0; t < NT; ++t) {
            half16 bm;
            #pragma unroll
            for (int h = 0; h < 16; ++h)
                bm[h] = Xb[(long long)(kk + (h & 7) + ((h & 8) << 1) + bofs) * Ncols + t * 16];
            acc[t] = __builtin_amdgcn_wmma_f32_16x16x32_f16(false, a, false, bm,
                                                            (short)0, acc[t], false, false);
        }
    }

    #pragma unroll
    for (int t = 0; t < NT; ++t) {
        float* ob = out + (long long)b * O * Ncols + (long long)tO * 16 * Ncols
                        + (tN0 + t) * 16 + lid;
        #pragma unroll
        for (int r = 0; r < 8; ++r) {
            int m = r + hi * 8;
            float v = acc[t][r];
            if (bias) v += bias[tO * 16 + m];
            ob[(long long)m * Ncols] = v;
        }
    }
}

// ---------------------------------------------------------------------------
// BN batch statistics: mean/var per output channel over (B, Ncols)
// ---------------------------------------------------------------------------
__global__ __launch_bounds__(256)
void bn_stats_kernel(const float* __restrict__ h, int O, int Ncols, int B,
                     float* __restrict__ mean, float* __restrict__ var)
{
    const int o = blockIdx.x;
    const int tid = threadIdx.x;
    const long long count = (long long)B * Ncols;
    float s = 0.f, s2 = 0.f;
    for (long long i = tid; i < count; i += 256) {
        long long bb = i / Ncols, n = i % Ncols;
        float v = h[bb * O * Ncols + (long long)o * Ncols + n];
        s += v; s2 += v * v;
    }
    __shared__ float ss[256], ss2[256];
    ss[tid] = s; ss2[tid] = s2;
    __syncthreads();
    for (int w = 128; w; w >>= 1) {
        if (tid < w) { ss[tid] += ss[tid + w]; ss2[tid] += ss2[tid + w]; }
        __syncthreads();
    }
    if (tid == 0) {
        float mu = ss[0] / (float)count;
        mean[o] = mu;
        var[o]  = ss2[0] / (float)count - mu * mu;
    }
}

// ---------------------------------------------------------------------------
// BN apply + LeakyReLU (optionally also emits f16 copy)
// ---------------------------------------------------------------------------
__global__
void bn_lrelu_kernel(const float* __restrict__ hin, float* __restrict__ hout,
                     _Float16* __restrict__ hout16,
                     const float* __restrict__ mean, const float* __restrict__ var,
                     const float* __restrict__ g, const float* __restrict__ bb,
                     int O, int Ncols, long long total)
{
    long long i = (long long)blockIdx.x * blockDim.x + threadIdx.x;
    if (i >= total) return;
    long long on = i % ((long long)O * Ncols);
    int o = (int)(on / Ncols);
    float v = (hin[i] - mean[o]) * rsqrtf(var[o] + BN_EPS) * g[o] + bb[o];
    v = lrelu(v);
    hout[i] = v;
    if (hout16) hout16[i] = (_Float16)v;
}

// ---------------------------------------------------------------------------
// graph_feature: out[:, c] = h[c,n]; out[:, C+c] = max_k h[c, idx[k]] - h[c,n]
// writes directly into the 640-channel concat buffer (f32 + f16 copies)
// ---------------------------------------------------------------------------
__global__ __launch_bounds__(128)
void graph_feature_kernel(const float* __restrict__ h, int C,
                          const int* __restrict__ idx,
                          float* __restrict__ xc, _Float16* __restrict__ xc16,
                          int chanOff)
{
    const int n = blockIdx.x;
    const int b = blockIdx.y;
    const float* hb = h + (long long)b * C * NPTS;
    __shared__ int sidx[KNN_K];
    if (threadIdx.x < KNN_K)
        sidx[threadIdx.x] = idx[((long long)b * NPTS + n) * KNN_K + threadIdx.x];
    __syncthreads();
    const long long obase = (long long)b * 640 * NPTS + n;
    for (int c = threadIdx.x; c < C; c += 128) {
        const float* hc = hb + (long long)c * NPTS;
        float f = hc[n];
        float m = NEGINF;
        #pragma unroll
        for (int k = 0; k < KNN_K; ++k) { float v = hc[sidx[k]]; m = v > m ? v : m; }
        long long o0 = obase + (long long)(chanOff + c) * NPTS;
        long long o1 = obase + (long long)(chanOff + C + c) * NPTS;
        xc[o0] = f;      xc16[o0] = (_Float16)f;
        xc[o1] = m - f;  xc16[o1] = (_Float16)(m - f);
    }
}

// ---------------------------------------------------------------------------
// feat[b][n] = max over 1024 channels of h5; store transposed f16 (1024 x 32)
// ---------------------------------------------------------------------------
__global__ __launch_bounds__(32)
void featmax_kernel(const float* __restrict__ h5, _Float16* __restrict__ featT)
{
    const int n = blockIdx.x;
    const int b = blockIdx.y;
    const float* p = h5 + (long long)b * 1024 * NPTS + n;
    float m = NEGINF;
    for (int c = threadIdx.x; c < 1024; c += 32) {
        float v = p[(long long)c * NPTS]; m = v > m ? v : m;
    }
    for (int off = 16; off; off >>= 1) {
        float ov = __shfl_xor(m, off, 32); m = ov > m ? ov : m;
    }
    if (threadIdx.x == 0) featT[(long long)n * 32 + b] = (_Float16)m;
}

// ---------------------------------------------------------------------------
// converts
// ---------------------------------------------------------------------------
__global__
void cvt_pad_kernel(const float* __restrict__ src, _Float16* __restrict__ dst,
                    int O, int C, int Kpad)
{
    long long i = (long long)blockIdx.x * blockDim.x + threadIdx.x;
    if (i >= (long long)O * Kpad) return;
    int o = (int)(i / Kpad), k = (int)(i % Kpad);
    dst[i] = (k < C) ? (_Float16)src[(long long)o * C + k] : (_Float16)0.f;
}

__global__
void cvt_xpad16_kernel(const float* __restrict__ x, _Float16* __restrict__ xp)
{
    long long i = (long long)blockIdx.x * blockDim.x + threadIdx.x; // B*32*NPTS
    if (i >= (long long)BATCH * 32 * NPTS) return;
    long long b = i / (32 * NPTS);
    int k = (int)((i / NPTS) % 32);
    int n = (int)(i % NPTS);
    xp[i] = (k < 3) ? (_Float16)x[(b * 3 + k) * NPTS + n] : (_Float16)0.f;
}

// f32 pad of xyz to C=4 so the knn WMMA loop needs no bounds checks
__global__
void cvt_xpad4f_kernel(const float* __restrict__ x, float* __restrict__ xp)
{
    long long i = (long long)blockIdx.x * blockDim.x + threadIdx.x; // B*4*NPTS
    if (i >= (long long)BATCH * 4 * NPTS) return;
    long long b = i / (4 * NPTS);
    int k = (int)((i / NPTS) & 3);
    int n = (int)(i % NPTS);
    xp[i] = (k < 3) ? x[(b * 3 + k) * NPTS + n] : 0.f;
}

// ---------------------------------------------------------------------------
// final FC3: out[s][o] = bl3[o] + sum_k act2[k][s] * L3[o][k]   (40x256, tiny)
// ---------------------------------------------------------------------------
__global__
void fc3_kernel(const float* __restrict__ a /*(256 x 32)*/,
                const float* __restrict__ L3, const float* __restrict__ bl3,
                float* __restrict__ out)
{
    int i = blockIdx.x * blockDim.x + threadIdx.x;
    if (i >= 32 * 40) return;
    int s = i / 40, o = i % 40;
    float acc = bl3[o];
    for (int k = 0; k < 256; ++k) acc += a[(long long)k * 32 + s] * L3[o * 256 + k];
    out[i] = acc;
}

// ---------------------------------------------------------------------------
// host orchestration
// ---------------------------------------------------------------------------
static inline size_t align256(size_t x) { return (x + 255) & ~(size_t)255; }

extern "C" void kernel_launch(void* const* d_in, const int* in_sizes, int n_in,
                              void* d_out, int out_size, void* d_ws, size_t ws_size,
                              hipStream_t stream)
{
    const float* x   = (const float*)d_in[0];
    const float* W1  = (const float*)d_in[1];  const float* g1 = (const float*)d_in[2];  const float* b1 = (const float*)d_in[3];
    const float* W2  = (const float*)d_in[4];  const float* g2 = (const float*)d_in[5];  const float* b2 = (const float*)d_in[6];
    const float* W3  = (const float*)d_in[7];  const float* g3 = (const float*)d_in[8];  const float* b3 = (const float*)d_in[9];
    const float* W4  = (const float*)d_in[10]; const float* g4 = (const float*)d_in[11]; const float* b4 = (const float*)d_in[12];
    const float* W5  = (const float*)d_in[13]; const float* g5 = (const float*)d_in[14]; const float* b5 = (const float*)d_in[15];
    const float* L1  = (const float*)d_in[16]; const float* g6 = (const float*)d_in[17]; const float* b6 = (const float*)d_in[18];
    const float* L2  = (const float*)d_in[19]; const float* bl2 = (const float*)d_in[20];
    const float* g7  = (const float*)d_in[21]; const float* b7 = (const float*)d_in[22];
    const float* L3  = (const float*)d_in[23]; const float* bl3 = (const float*)d_in[24];

    char* w = (char*)d_ws;
    size_t off = 0;
    auto alloc = [&](size_t bytes) -> void* {
        void* p = w + off; off = align256(off + bytes); return p;
    };

    float*     pd      = (float*)    alloc((size_t)BATCH * NPTS * NPTS * 4);   // 128 MB (aliased by h5)
    int*       idx     = (int*)      alloc((size_t)BATCH * NPTS * KNN_K * 4);
    float*     x0pad   = (float*)    alloc((size_t)BATCH * 4 * NPTS * 4);      // xyz padded to C=4 (f32)
    _Float16*  xpad16  = (_Float16*) alloc((size_t)BATCH * 32 * NPTS * 2);
    _Float16*  w1_16   = (_Float16*) alloc((size_t)64 * 32 * 2);
    _Float16*  w2_16   = (_Float16*) alloc((size_t)64 * 128 * 2);
    _Float16*  w3_16   = (_Float16*) alloc((size_t)64 * 128 * 2);
    _Float16*  w4_16   = (_Float16*) alloc((size_t)128 * 128 * 2);
    _Float16*  w5_16   = (_Float16*) alloc((size_t)1024 * 640 * 2);
    _Float16*  l1_16   = (_Float16*) alloc((size_t)512 * 1024 * 2);
    _Float16*  l2_16   = (_Float16*) alloc((size_t)256 * 512 * 2);
    float*     hraw    = (float*)    alloc((size_t)BATCH * 128 * NPTS * 4);    // conv1-4 pre-act
    float*     hact    = (float*)    alloc((size_t)BATCH * 128 * NPTS * 4);    // conv1-4 act
    float*     xc      = (float*)    alloc((size_t)BATCH * 640 * NPTS * 4);    // concat f32
    _Float16*  xc16    = (_Float16*) alloc((size_t)BATCH * 640 * NPTS * 2);    // concat f16
    float*     meanb   = (float*)    alloc(1024 * 4);
    float*     varb    = (float*)    alloc(1024 * 4);
    _Float16*  featT   = (_Float16*) alloc((size_t)1024 * 32 * 2);
    float*     hraw1   = (float*)    alloc((size_t)512 * 32 * 4);
    float*     act1    = (float*)    alloc((size_t)512 * 32 * 4);
    _Float16*  act1_16 = (_Float16*) alloc((size_t)512 * 32 * 2);
    float*     hraw2   = (float*)    alloc((size_t)256 * 32 * 4);
    float*     act2    = (float*)    alloc((size_t)256 * 32 * 4);
    (void)ws_size; (void)n_in; (void)in_sizes; (void)out_size;

    // ---- converts -------------------------------------------------------
    {
        long long n16 = (long long)BATCH * 32 * NPTS;
        cvt_xpad16_kernel<<<(unsigned)((n16 + 255) / 256), 256, 0, stream>>>(x, xpad16);
        long long n4 = (long long)BATCH * 4 * NPTS;
        cvt_xpad4f_kernel<<<(unsigned)((n4 + 255) / 256), 256, 0, stream>>>(x, x0pad);
        auto cvt = [&](const float* s, _Float16* d, int O, int C, int Kp) {
            long long t = (long long)O * Kp;
            cvt_pad_kernel<<<(unsigned)((t + 255) / 256), 256, 0, stream>>>(s, d, O, C, Kp);
        };
        cvt(W1, w1_16, 64, 3, 32);
        cvt(W2, w2_16, 64, 128, 128);
        cvt(W3, w3_16, 64, 128, 128);
        cvt(W4, w4_16, 128, 128, 128);
        cvt(W5, w5_16, 1024, 640, 640);
        cvt(L1, l1_16, 512, 1024, 1024);
        cvt(L2, l2_16, 256, 512, 512);
    }

    auto knn_stage = [&](const float* src, int chanOff, int C, long long bS) {
        knn_pd_kernel<4><<<dim3(NPTS / 64, NPTS / 16, BATCH), 32, 0, stream>>>(
            src, chanOff, C, bS, pd);
        topk_kernel<<<BATCH * NPTS, 32, 0, stream>>>(pd, idx);
    };
    auto conv_stage = [&](const _Float16* Wf, int Kdim, const _Float16* Xf, int chanOff,
                          long long xbS, int O, const float* g, const float* bb) {
        gemm_f16_kernel<4><<<dim3(NPTS / 64, O / 16, BATCH), 32, 0, stream>>>(
            Wf, Kdim, Xf, chanOff, xbS, nullptr, hraw, O, NPTS);
        bn_stats_kernel<<<O, 256, 0, stream>>>(hraw, O, NPTS, BATCH, meanb, varb);
        long long total = (long long)BATCH * O * NPTS;
        bn_lrelu_kernel<<<(unsigned)((total + 255) / 256), 256, 0, stream>>>(
            hraw, hact, (_Float16*)nullptr, meanb, varb, g, bb, O, NPTS, total);
    };
    auto gf = [&](int C, int chanOff) {
        graph_feature_kernel<<<dim3(NPTS, BATCH), 128, 0, stream>>>(hact, C, idx, xc, xc16, chanOff);
    };

    const long long xcS = (long long)640 * NPTS;

    // stage 1: knn on padded xyz (C=4, last channel zero -> same distances)
    knn_stage(x0pad, 0, 4, (long long)4 * NPTS);
    conv_stage(w1_16, 32, xpad16, 0, (long long)32 * NPTS, 64, g1, b1);
    gf(64, 0);                                    // x1 -> channels 0..127

    // stage 2
    knn_stage(xc, 0, 128, xcS);
    conv_stage(w2_16, 128, xc16, 0, xcS, 64, g2, b2);
    gf(64, 128);                                  // x2 -> channels 128..255

    // stage 3
    knn_stage(xc, 128, 128, xcS);
    conv_stage(w3_16, 128, xc16, 128, xcS, 64, g3, b3);
    gf(64, 256);                                  // x3 -> channels 256..383

    // stage 4
    knn_stage(xc, 256, 128, xcS);
    conv_stage(w4_16, 128, xc16, 256, xcS, 128, g4, b4);
    gf(128, 384);                                 // x4 -> channels 384..639

    // conv5 (640 -> 1024) + BN + lrelu + channel-max ; h5 aliases pd (dead)
    float* h5 = pd;
    gemm_f16_kernel<4><<<dim3(NPTS / 64, 1024 / 16, BATCH), 32, 0, stream>>>(
        w5_16, 640, xc16, 0, xcS, nullptr, h5, 1024, NPTS);
    bn_stats_kernel<<<1024, 256, 0, stream>>>(h5, 1024, NPTS, BATCH, meanb, varb);
    {
        long long total = (long long)BATCH * 1024 * NPTS;
        bn_lrelu_kernel<<<(unsigned)((total + 255) / 256), 256, 0, stream>>>(
            h5, h5, (_Float16*)nullptr, meanb, varb, g5, b5, 1024, NPTS, total);
    }
    featmax_kernel<<<dim3(NPTS, BATCH), 32, 0, stream>>>(h5, featT);

    // fc1: (512 x 1024) * featT(1024 x 32)
    gemm_f16_kernel<2><<<dim3(1, 512 / 16, 1), 32, 0, stream>>>(
        l1_16, 1024, featT, 0, 0, nullptr, hraw1, 512, 32);
    bn_stats_kernel<<<512, 256, 0, stream>>>(hraw1, 512, 32, 1, meanb, varb);
    bn_lrelu_kernel<<<(512 * 32 + 255) / 256, 256, 0, stream>>>(
        hraw1, act1, act1_16, meanb, varb, g6, b6, 512, 32, 512 * 32);

    // fc2: (256 x 512) * act1(512 x 32) + bl2
    gemm_f16_kernel<2><<<dim3(1, 256 / 16, 1), 32, 0, stream>>>(
        l2_16, 512, act1_16, 0, 0, bl2, hraw2, 256, 32);
    bn_stats_kernel<<<256, 256, 0, stream>>>(hraw2, 256, 32, 1, meanb, varb);
    bn_lrelu_kernel<<<(256 * 32 + 255) / 256, 256, 0, stream>>>(
        hraw2, act2, (_Float16*)nullptr, meanb, varb, g7, b7, 256, 32, 256 * 32);

    // fc3 -> d_out (32 x 40)
    fc3_kernel<<<(32 * 40 + 127) / 128, 128, 0, stream>>>(act2, L3, bl3, (float*)d_out);
}
